// AMPBlock0_58128087384831
// MI455X (gfx1250) — matrix operand order repaired and checked by reading, not compiled, not run on hardware
//
#include <hip/hip_runtime.h>
#include <hip/hip_bf16.h>

#define Cc 512
#define Tt 8192
#define Bb 8
#define ROWS 520   // LDS row stride in ushorts (1040 B, 16B-aligned, bank-skewed)

typedef __attribute__((ext_vector_type(16))) __bf16 v16bf;
typedef __attribute__((ext_vector_type(8)))  float  v8f;

union BFrag { uint4 q[2]; v16bf v; };

__device__ __forceinline__ unsigned short f2bf(float f) {
    union { float f; unsigned int u; } v; v.f = f;
    unsigned int u = v.u;
    unsigned int r = u + 0x7fffu + ((u >> 16) & 1u);
    return (unsigned short)(r >> 16);
}

// snake with precomputed ta = 2*exp(alpha), rb = 1/(2*exp(beta)+eps)
__device__ __forceinline__ float snakef(float u, float ta, float rb) {
    return u + (1.0f - __cosf(ta * u)) * rb;
}

// ---------------------------------------------------------------------------
// prep: weight-norm, bf16 pack into WMMA A-fragment order, exp params
// blocks 0..511 -> w1 (co=bid), 512..1023 -> w2
// ---------------------------------------------------------------------------
__global__ void prep_kernel(const float* __restrict__ v1, const float* __restrict__ g1,
                            const float* __restrict__ v2, const float* __restrict__ g2,
                            const float* __restrict__ a1, const float* __restrict__ b1,
                            const float* __restrict__ a2, const float* __restrict__ b2,
                            unsigned short* __restrict__ w1p,
                            unsigned short* __restrict__ w2p,
                            float* __restrict__ expp) {
    const int bid = blockIdx.x;
    const float* v; const float* g; unsigned short* wp; int co;
    if (bid < Cc) { v = v1; g = g1; wp = w1p; co = bid; }
    else          { v = v2; g = g2; wp = w2p; co = bid - Cc; }

    __shared__ float red[256];
    float s = 0.0f;
    for (int i = threadIdx.x; i < Cc * 3; i += 256) {
        float w = v[co * (Cc * 3) + i];
        s += w * w;
    }
    red[threadIdx.x] = s;
    __syncthreads();
    for (int off = 128; off > 0; off >>= 1) {
        if ((int)threadIdx.x < off) red[threadIdx.x] += red[threadIdx.x + off];
        __syncthreads();
    }
    const float sc = g[co] / sqrtf(red[0]);

    const int cb = co >> 4, m = co & 15;
    for (int kd = threadIdx.x; kd < Cc * 3; kd += 256) {
        int k  = kd / Cc;        // tap 0..2
        int ci = kd % Cc;
        int q  = k * 16 + (ci >> 5);
        int K  = ci & 31;
        int lane, j;
        if      (K < 8)  { lane = m;      j = K;      }
        else if (K < 16) { lane = m + 16; j = K - 8;  }
        else if (K < 24) { lane = m;      j = K - 8;  }
        else             { lane = m + 16; j = K - 16; }
        float w = sc * v[(co * Cc + ci) * 3 + k];
        wp[((size_t)(cb * 48 + q)) * 512 + lane * 16 + j] = f2bf(w);
    }

    if (bid == 0) {
        for (int i = threadIdx.x; i < Cc; i += 256) {
            expp[i]          = 2.0f * __expf(a1[i]);
            expp[Cc + i]     = 1.0f / (2.0f * __expf(b1[i]) + 1e-9f);
            expp[2 * Cc + i] = 2.0f * __expf(a2[i]);
            expp[3 * Cc + i] = 1.0f / (2.0f * __expf(b2[i]) + 1e-9f);
        }
    }
}

// ---------------------------------------------------------------------------
// act1: x [B,C,T] f32 (t contiguous) -> actT [B][T+2][C] bf16 (zero pad rows)
// ---------------------------------------------------------------------------
__global__ void act1_kernel(const float* __restrict__ x,
                            const float* __restrict__ expp,
                            unsigned short* __restrict__ actT) {
    const int b  = blockIdx.z;
    const int cb = blockIdx.y * 32;
    const int tb = blockIdx.x * 32;
    const int tx = threadIdx.x & 31;
    const int ty = threadIdx.x >> 5;

    __shared__ float xs[32][34];
    __shared__ float ys[32][33];

    #pragma unroll
    for (int i = 0; i < 4; ++i) {
        int cl = ty + 8 * i;
        const float* row = x + ((size_t)b * Cc + cb + cl) * Tt;
        int tg = tb - 1 + tx; tg = min(max(tg, 0), Tt - 1);
        xs[cl][tx] = row[tg];
        if (tx < 2) {
            int tg2 = tb + 31 + tx; tg2 = min(max(tg2, 0), Tt - 1);
            xs[cl][32 + tx] = row[tg2];
        }
    }
    __syncthreads();

    #pragma unroll
    for (int i = 0; i < 4; ++i) {
        int cl = ty + 8 * i;
        float ta = expp[cb + cl];
        float rb = expp[Cc + cb + cl];
        float xm = xs[cl][tx], x0 = xs[cl][tx + 1], xp = xs[cl][tx + 2];
        float u0 = 0.25f * xm + 0.75f * x0;   // up[2t]   (clamped halo == torch clamp)
        float u1 = 0.75f * x0 + 0.25f * xp;   // up[2t+1]
        ys[tx][cl] = 0.5f * (snakef(u0, ta, rb) + snakef(u1, ta, rb));
    }
    __syncthreads();

    #pragma unroll
    for (int i = 0; i < 4; ++i) {
        int tr = ty + 8 * i;
        actT[((size_t)b * (Tt + 2) + tb + tr + 1) * Cc + cb + tx] = f2bf(ys[tr][tx]);
    }
    if (tb == 0 && ty == 0)
        actT[((size_t)b * (Tt + 2)) * Cc + cb + tx] = 0;
    if (tb == Tt - 32 && ty == 0)
        actT[((size_t)b * (Tt + 2) + Tt + 1) * Cc + cb + tx] = 0;
}

// ---------------------------------------------------------------------------
// act2: z1T [B][T][C] f32 (c contiguous) -> actT [B][T+2][C] bf16  (streaming)
// ---------------------------------------------------------------------------
__global__ void act2_kernel(const float* __restrict__ z,
                            const float* __restrict__ expp,
                            unsigned short* __restrict__ actT) {
    const int b  = blockIdx.y;
    const int t  = blockIdx.x * 2 + (threadIdx.x >> 7);
    const int c4 = threadIdx.x & 127;

    const float4* Z = (const float4*)(z + (size_t)b * Tt * Cc);
    const int tm = max(t - 1, 0), tp = min(t + 1, Tt - 1);
    float4 zm = Z[(size_t)tm * 128 + c4];
    float4 z0 = Z[(size_t)t  * 128 + c4];
    float4 zp = Z[(size_t)tp * 128 + c4];
    float4 ta = ((const float4*)(expp + 2 * Cc))[c4];
    float4 rb = ((const float4*)(expp + 3 * Cc))[c4];

    float ym[4], y0[4], yp[4], a[4], r[4], o[4];
    ym[0]=zm.x; ym[1]=zm.y; ym[2]=zm.z; ym[3]=zm.w;
    y0[0]=z0.x; y0[1]=z0.y; y0[2]=z0.z; y0[3]=z0.w;
    yp[0]=zp.x; yp[1]=zp.y; yp[2]=zp.z; yp[3]=zp.w;
    a[0]=ta.x; a[1]=ta.y; a[2]=ta.z; a[3]=ta.w;
    r[0]=rb.x; r[1]=rb.y; r[2]=rb.z; r[3]=rb.w;

    #pragma unroll
    for (int i = 0; i < 4; ++i) {
        float u0 = 0.25f * ym[i] + 0.75f * y0[i];
        float u1 = 0.75f * y0[i] + 0.25f * yp[i];
        o[i] = 0.5f * (snakef(u0, a[i], r[i]) + snakef(u1, a[i], r[i]));
    }
    unsigned int pk0 = (unsigned int)f2bf(o[0]) | ((unsigned int)f2bf(o[1]) << 16);
    unsigned int pk1 = (unsigned int)f2bf(o[2]) | ((unsigned int)f2bf(o[3]) << 16);

    unsigned int* dst = (unsigned int*)(actT + ((size_t)b * (Tt + 2) + t + 1) * Cc + c4 * 4);
    dst[0] = pk0; dst[1] = pk1;

    if (blockIdx.x == 0 && threadIdx.x < 128) {
        unsigned int* p0 = (unsigned int*)(actT + ((size_t)b * (Tt + 2)) * Cc + c4 * 4);
        p0[0] = 0u; p0[1] = 0u;
    }
    if (blockIdx.x == Tt / 2 - 1 && threadIdx.x < 128) {
        unsigned int* p1 = (unsigned int*)(actT + ((size_t)b * (Tt + 2) + Tt + 1) * Cc + c4 * 4);
        p1[0] = 0u; p1[1] = 0u;
    }
}

// ---------------------------------------------------------------------------
// conv_gemm v2: block = 32-wide t tile, async-staged activation tile in LDS,
// each wave does 2 passes of a 2x2 (co x t) WMMA register tile:
//   per K-chunk: 2 A-frags (global b128) + 2 B-frags (ds b128) -> 4 WMMAs.
// FINAL=false: bias, store f32 t-major z1T [B][T][C] (8 contiguous floats/lane)
// FINAL=true : bias + LDS transpose + residual, store [B,C,T] to d_out
// ---------------------------------------------------------------------------
template <bool FINAL>
__global__ void conv_gemm_kernel(const unsigned short* __restrict__ actT,
                                 const unsigned short* __restrict__ wp,
                                 const float* __restrict__ bias,
                                 const float* __restrict__ resid,
                                 float* __restrict__ out) {
    const int b     = blockIdx.y;
    const int tbase = blockIdx.x * 32;
    const int wave  = threadIdx.x >> 5;
    const int lane  = threadIdx.x & 31;
    const int half  = lane >> 4;
    const int n     = lane & 15;

    __shared__ __align__(16) unsigned short sB[34 * ROWS];

    const unsigned short* act_b = actT + (size_t)b * (Tt + 2) * Cc;

    // ---- async-stage activation rows [tbase .. tbase+33] x [0..511] to LDS ----
    for (int idx = threadIdx.x; idx < 34 * 64; idx += 256) {
        const int row = idx >> 6;
        const int col = (idx & 63) * 8;                      // ushort units (16B)
        const unsigned short* g = act_b + (size_t)(tbase + row) * Cc + col;
        unsigned int laddr = (unsigned int)(size_t)&sB[row * ROWS + col];
        asm volatile("global_load_async_to_lds_b128 %0, %1, off"
                     :: "v"(laddr), "v"(g) : "memory");
    }
    asm volatile("s_wait_asynccnt 0x0" ::: "memory");
    __syncthreads();

    for (int itp = 0; itp < 2; ++itp) {
        const int pi  = wave + 8 * itp;                 // co pair 0..15
        const int cb0 = pi * 2;
        const int cb1 = cb0 + 1;
        v8f acc[2][2] = {};
        const unsigned short* wb0 = wp + (size_t)cb0 * 48 * 512 + lane * 16;
        const unsigned short* wb1 = wp + (size_t)cb1 * 48 * 512 + lane * 16;

        #pragma unroll
        for (int k = 0; k < 3; ++k) {
            const unsigned short* l0 = &sB[(n + k) * ROWS + half * 8];       // t-sub 0
            const unsigned short* l1 = &sB[(n + 16 + k) * ROWS + half * 8];  // t-sub 1
            #pragma unroll 2
            for (int cib = 0; cib < 16; ++cib) {
                const int q = k * 16 + cib;
                BFrag A0, A1, B0, B1;
                const uint4* a0 = (const uint4*)(wb0 + (size_t)q * 512);
                const uint4* a1 = (const uint4*)(wb1 + (size_t)q * 512);
                A0.q[0] = a0[0]; A0.q[1] = a0[1];
                A1.q[0] = a1[0]; A1.q[1] = a1[1];
                const uint4* p0 = (const uint4*)(l0 + cib * 32);
                const uint4* p1 = (const uint4*)(l1 + cib * 32);
                B0.q[0] = p0[0]; B0.q[1] = p0[2];
                B1.q[0] = p1[0]; B1.q[1] = p1[2];
                acc[0][0] = __builtin_amdgcn_wmma_f32_16x16x32_bf16(
                                false, A0.v, false, B0.v, (short)0, acc[0][0], false, false);
                acc[0][1] = __builtin_amdgcn_wmma_f32_16x16x32_bf16(
                                false, A0.v, false, B1.v, (short)0, acc[0][1], false, false);
                acc[1][0] = __builtin_amdgcn_wmma_f32_16x16x32_bf16(
                                false, A1.v, false, B0.v, (short)0, acc[1][0], false, false);
                acc[1][1] = __builtin_amdgcn_wmma_f32_16x16x32_bf16(
                                false, A1.v, false, B1.v, (short)0, acc[1][1], false, false);
            }
        }

        const int cbs[2] = { cb0, cb1 };
        float bv[2][8];
        #pragma unroll
        for (int p = 0; p < 2; ++p) {
            const int cobase = cbs[p] * 16 + half * 8;
            #pragma unroll
            for (int r = 0; r < 8; ++r) bv[p][r] = bias[cobase + r];
        }

        if constexpr (!FINAL) {
            #pragma unroll
            for (int p = 0; p < 2; ++p) {
                const int cobase = cbs[p] * 16 + half * 8;
                #pragma unroll
                for (int s = 0; s < 2; ++s) {
                    float* orow = out + ((size_t)b * Tt + tbase + s * 16 + n) * Cc + cobase;
                    float4 lo = { acc[p][s][0] + bv[p][0], acc[p][s][1] + bv[p][1],
                                  acc[p][s][2] + bv[p][2], acc[p][s][3] + bv[p][3] };
                    float4 hi = { acc[p][s][4] + bv[p][4], acc[p][s][5] + bv[p][5],
                                  acc[p][s][6] + bv[p][6], acc[p][s][7] + bv[p][7] };
                    ((float4*)orow)[0] = lo;
                    ((float4*)orow)[1] = hi;
                }
            }
        } else {
            __shared__ float xpose[8][16][17];
            #pragma unroll
            for (int p = 0; p < 2; ++p) {
                #pragma unroll
                for (int s = 0; s < 2; ++s) {
                    __syncthreads();   // uniform: all waves run identical loop structure
                    #pragma unroll
                    for (int r = 0; r < 8; ++r)
                        xpose[wave][half * 8 + r][n] = acc[p][s][r] + bv[p][r];
                    __syncthreads();
                    const int m2 = lane >> 1, tseg = (lane & 1) * 8;
                    const int co = cbs[p] * 16 + m2;
                    const size_t base = ((size_t)b * Cc + co) * Tt + tbase + s * 16 + tseg;
                    float4 r0 = ((const float4*)(resid + base))[0];
                    float4 r1 = ((const float4*)(resid + base))[1];
                    float4 o0 = { xpose[wave][m2][tseg + 0] + r0.x,
                                  xpose[wave][m2][tseg + 1] + r0.y,
                                  xpose[wave][m2][tseg + 2] + r0.z,
                                  xpose[wave][m2][tseg + 3] + r0.w };
                    float4 o1 = { xpose[wave][m2][tseg + 4] + r1.x,
                                  xpose[wave][m2][tseg + 5] + r1.y,
                                  xpose[wave][m2][tseg + 6] + r1.z,
                                  xpose[wave][m2][tseg + 7] + r1.w };
                    ((float4*)(out + base))[0] = o0;
                    ((float4*)(out + base))[1] = o1;
                }
            }
        }
    }
}

// ---------------------------------------------------------------------------
extern "C" void kernel_launch(void* const* d_in, const int* in_sizes, int n_in,
                              void* d_out, int out_size, void* d_ws, size_t ws_size,
                              hipStream_t stream) {
    (void)in_sizes; (void)n_in; (void)out_size; (void)ws_size;
    const float* x      = (const float*)d_in[0];
    const float* v1     = (const float*)d_in[1];
    const float* g1     = (const float*)d_in[2];
    const float* bias1  = (const float*)d_in[3];
    const float* v2     = (const float*)d_in[4];
    const float* g2     = (const float*)d_in[5];
    const float* bias2  = (const float*)d_in[6];
    const float* alpha1 = (const float*)d_in[7];
    const float* beta1  = (const float*)d_in[8];
    const float* alpha2 = (const float*)d_in[9];
    const float* beta2  = (const float*)d_in[10];

    char* ws = (char*)d_ws;
    unsigned short* w1p  = (unsigned short*)(ws);                 // 1.5 MB
    unsigned short* w2p  = (unsigned short*)(ws + 0x180000);      // 1.5 MB
    float*          expp = (float*)(ws + 0x300000);               // 8 KB
    unsigned short* actT = (unsigned short*)(ws + 0x400000);      // 67 MB (reused)
    float*          z1   = (float*)(ws + 0x4800000);              // 128 MB

    prep_kernel<<<2 * Cc, 256, 0, stream>>>(v1, g1, v2, g2, alpha1, beta1,
                                            alpha2, beta2, w1p, w2p, expp);
    act1_kernel<<<dim3(Tt / 32, Cc / 32, Bb), 256, 0, stream>>>(x, expp, actT);
    conv_gemm_kernel<false><<<dim3(Tt / 32, Bb), 256, 0, stream>>>(
        actT, w1p, bias1, nullptr, z1);
    act2_kernel<<<dim3(Tt / 2, Bb), 256, 0, stream>>>(z1, expp, actT);
    conv_gemm_kernel<true><<<dim3(Tt / 32, Bb), 256, 0, stream>>>(
        actT, w2p, bias2, x, (float*)d_out);
}
